// EMLSparseRouter_3332894621796
// MI455X (gfx1250) — matrix operand order repaired and checked by reading, compiled, not run on hardware
//
#include <hip/hip_runtime.h>
#include <hip/hip_bf16.h>

// ---------------------------------------------------------------------------
// EML sparse router, fully fused for gfx1250 (MI455X).
// Shapes (fixed by reference): B=64, S=4096, D=128, E=256, H=256, K_top=8.
// ---------------------------------------------------------------------------

typedef __bf16 bf16_t;
typedef __attribute__((ext_vector_type(16))) __bf16 v16bf;
typedef __attribute__((ext_vector_type(8)))  float  v8f;

union AFrag { v16bf v; uint4 u[2]; };

// ---- output layout (flat, reference return order) -------------------------
#define EN_OFF   0u              // energy        [64,4096]
#define GT_OFF   262144u         // gate          [64,4096]
#define DR_OFF   524288u         // drive         [64,4096]
#define RS_OFF   786432u         // resistance    [64,4096]
#define IDX_OFF  1048576u        // topk_indices  [64,8]   (as float)
#define SC_OFF   1049088u        // topk_scores   [64,8]
#define MASK_OFF 1049600u        // active_mask   [64,4096]
#define AS_OFF   1311744u        // active_states [64,8,128]

__device__ __forceinline__ float gelu_tanh(float x) {
  float x3 = x * x * x;
  return 0.5f * x * (1.f + tanhf(0.7978845608028654f * (x + 0.044715f * x3)));
}

__device__ __forceinline__ v8f wmma_bf16(v16bf a, v16bf b, v8f c) {
  // D = A(16x32 bf16) * B(32x16 bf16) + C(16x16 f32)
  return __builtin_amdgcn_wmma_f32_16x16x32_bf16(
      /*neg_a=*/false, a, /*neg_b=*/false, b,
      /*c_mod=*/(short)0, c, /*reuse_a=*/false, /*reuse_b=*/false);
}

__device__ __forceinline__ v16bf load_frag16(const bf16_t* p) {
  AFrag f;
  f.u[0] = *(const uint4*)p;        // 8 halves (16B)
  f.u[1] = *(const uint4*)(p + 16); // +16 halves for A; caller offsets for B
  return f.v;
}

__device__ __forceinline__ v16bf load_fragB(const bf16_t* p) {
  AFrag f;
  const uint4* q = (const uint4*)p; // 16 contiguous halves (32B)
  f.u[0] = q[0];
  f.u[1] = q[1];
  return f.v;
}

// ---------------------------------------------------------------------------
// Prep: f32 -> bf16 weight conversion (weights stay L2-resident, ~0.5 MB)
// ---------------------------------------------------------------------------
__global__ void cvt_f32_bf16(const float* __restrict__ src,
                             bf16_t* __restrict__ dst, int n) {
  int i = blockIdx.x * blockDim.x + threadIdx.x;
  if (i < n) dst[i] = (bf16_t)src[i];
}

// event_features[64,128] = event @ ep_W^T + ep_b   (tiny: plain FMA)
__global__ void event_proj_kernel(const float* __restrict__ event,
                                  const float* __restrict__ ep_W,
                                  const float* __restrict__ ep_b,
                                  float* __restrict__ ef) {
  int id = blockIdx.x * 256 + threadIdx.x;   // 8192 = 64*128
  int b = id >> 7, d = id & 127;
  const float* ev = event + b * 256;
  const float* w  = ep_W + d * 256;
  float acc = ep_b[d];
#pragma unroll 8
  for (int e = 0; e < 256; ++e) acc = fmaf(ev[e], w[e], acc);
  ef[id] = acc;
}

// ---------------------------------------------------------------------------
// Main fused kernel: one block (8 waves) per (batch, 32-slot tile).
// GEMM1 (WMMA): slot[32x128] @ sp_W^T -> sf
// LN over concat(ef, sf, ef*sf) -> joint bf16 in LDS
// GEMM2 (WMMA): joint[32x384] @ W1cat^T[512x384] -> h, fused gelu + dot(W2)
// ---------------------------------------------------------------------------
__global__ void __launch_bounds__(256)
eml_main_kernel(const float* __restrict__ slot_states,
                const float* __restrict__ sp_b,
                const float* __restrict__ ln_g,
                const float* __restrict__ ln_b,
                const float* __restrict__ d_b1,
                const float* __restrict__ d_W2,
                const float* __restrict__ d_b2,
                const float* __restrict__ r_b1,
                const float* __restrict__ r_W2,
                const float* __restrict__ r_b2,
                const float* __restrict__ gate_bias,
                const bf16_t* __restrict__ wSpW,   // [128][128] bf16
                const bf16_t* __restrict__ wW1,    // [512][384] bf16 (d rows, then r rows)
                const float*  __restrict__ wEF,    // [64][128] f32
                float* __restrict__ out) {
  __shared__ __align__(16) bf16_t sA[32][136];   // slot tile bf16 (pad: no bank conflict)
  __shared__ __align__(16) bf16_t sJ[32][392];   // joint bf16    (pad: no bank conflict)
  __shared__ float sSF[32][132];                 // slot_features f32
  __shared__ float sEF[128];
  __shared__ float sLNg[384], sLNb[384];
  __shared__ float sDrv[32], sRes[32];

  const int tid  = threadIdx.x;
  const int wave = tid >> 5;
  const int lane = tid & 31;
  const int half = lane >> 4;
  const int l16  = lane & 15;
  const int b    = blockIdx.x >> 7;          // 128 tiles per batch
  const int s0   = (blockIdx.x & 127) * 32;

  // ---- phase 0: stage ef, LN params, slot tile (f32 -> bf16), zero accs ----
  if (tid < 128) sEF[tid] = wEF[b * 128 + tid];
  for (int i = tid; i < 384; i += 256) { sLNg[i] = ln_g[i]; sLNb[i] = ln_b[i]; }
  {
    const float* src = slot_states + ((size_t)b * 4096 + s0) * 128;
#pragma unroll
    for (int i = 0; i < 16; ++i) {
      int e = tid + i * 256;                 // 4096 elements = 32x128
      int r = e >> 7, c = e & 127;
      sA[r][c] = (bf16_t)src[e];
    }
  }
  if (tid < 32) { sDrv[tid] = 0.f; sRes[tid] = 0.f; }
  __syncthreads();

  // ---- GEMM1: sf = slot @ sp_W^T + sp_b  (wave w owns column tile w) ------
  {
    v8f acc0 = {}, acc1 = {};
#pragma unroll
    for (int k0 = 0; k0 < 128; k0 += 32) {
      v16bf a0 = load_frag16(&sA[l16][k0 + 8 * half]);
      v16bf a1 = load_frag16(&sA[16 + l16][k0 + 8 * half]);
      v16bf bb = load_frag16(wSpW + (size_t)(wave * 16 + l16) * 128 + k0 + 8 * half);
      acc0 = wmma_bf16(a0, bb, acc0);
      acc1 = wmma_bf16(a1, bb, acc1);
    }
    const int col = wave * 16 + l16;
    const float bias = sp_b[col];
#pragma unroll
    for (int r = 0; r < 8; ++r) {
      int row = r + 8 * half;                // D layout: M = r + 8*(lane>>4)
      sSF[row][col]      = acc0[r] + bias;
      sSF[16 + row][col] = acc1[r] + bias;
    }
  }
  __syncthreads();

  // ---- LayerNorm over 384-wide concat [ef, sf, ef*sf]; wave per row -------
#pragma unroll
  for (int rr = 0; rr < 4; ++rr) {
    int row = wave * 4 + rr;
    float e[4], f[4], p[4], s = 0.f, s2 = 0.f;
#pragma unroll
    for (int i = 0; i < 4; ++i) {
      int j = lane + 32 * i;
      e[i] = sEF[j]; f[i] = sSF[row][j]; p[i] = e[i] * f[i];
      s  += e[i] + f[i] + p[i];
      s2 += e[i] * e[i] + f[i] * f[i] + p[i] * p[i];
    }
#pragma unroll
    for (int off = 16; off >= 1; off >>= 1) {
      s  += __shfl_xor(s, off, 32);
      s2 += __shfl_xor(s2, off, 32);
    }
    float mean = s * (1.f / 384.f);
    float var  = s2 * (1.f / 384.f) - mean * mean;
    float rstd = rsqrtf(var + 1e-5f);
#pragma unroll
    for (int i = 0; i < 4; ++i) {
      int j = lane + 32 * i;
      sJ[row][j]       = (bf16_t)(((e[i] - mean) * rstd) * sLNg[j]       + sLNb[j]);
      sJ[row][128 + j] = (bf16_t)(((f[i] - mean) * rstd) * sLNg[128 + j] + sLNb[128 + j]);
      sJ[row][256 + j] = (bf16_t)(((p[i] - mean) * rstd) * sLNg[256 + j] + sLNb[256 + j]);
    }
  }
  __syncthreads();

  // ---- GEMM2: h = joint @ W1cat^T ; fused gelu + dot(W2) epilogue ---------
  {
    v8f acc[8];
#pragma unroll
    for (int i = 0; i < 8; ++i) acc[i] = (v8f){};
    for (int k0 = 0; k0 < 384; k0 += 32) {
      v16bf a0 = load_frag16(&sJ[l16][k0 + 8 * half]);
      v16bf a1 = load_frag16(&sJ[16 + l16][k0 + 8 * half]);
#pragma unroll
      for (int i = 0; i < 4; ++i) {
        int ct = wave + 8 * i;               // column tile of the 512 hidden units
        v16bf bb = load_frag16(wW1 + (size_t)(ct * 16 + l16) * 384 + k0 + 8 * half);
        acc[2 * i]     = wmma_bf16(a0, bb, acc[2 * i]);
        acc[2 * i + 1] = wmma_bf16(a1, bb, acc[2 * i + 1]);
      }
    }
    // Epilogue: bias -> gelu -> *W2[n] -> 16-lane reduce -> LDS atomic add
#pragma unroll
    for (int i = 0; i < 4; ++i) {
      int n = (wave + 8 * i) * 16 + l16;     // hidden col 0..511 (tile never straddles 256)
      bool isD = (n < 256);
      float b1 = isD ? d_b1[n] : r_b1[n - 256];
      float w2 = isD ? d_W2[n] : r_W2[n - 256];
#pragma unroll
      for (int rt = 0; rt < 2; ++rt) {
        v8f a = acc[2 * i + rt];
        float c[8];
#pragma unroll
        for (int r = 0; r < 8; ++r) c[r] = gelu_tanh(a[r] + b1) * w2;
#pragma unroll
        for (int off = 1; off <= 8; off <<= 1)
#pragma unroll
          for (int r = 0; r < 8; ++r) c[r] += __shfl_xor(c[r], off, 32);
        if (l16 == 0) {                       // lanes 0 (rows r) and 16 (rows r+8)
          float* dst = isD ? sDrv : sRes;
#pragma unroll
          for (int r = 0; r < 8; ++r)
            atomicAdd(&dst[rt * 16 + r + 8 * half], c[r]);
        }
      }
    }
  }
  __syncthreads();

  // ---- finalize gate math, write the four [B,S] outputs -------------------
  if (tid < 32) {
    int row = tid;
    float dr = sDrv[row] + d_b2[0];
    float rw = sRes[row] + r_b2[0];
    float sp = (rw > 20.f) ? rw : log1pf(__expf(rw));     // softplus
    float en = dr / (sp + 1.f);
    en = fminf(3.f, fmaxf(-3.f, en));                     // CLIP_VALUE
    float gt = tanhf(en + gate_bias[0]);                  // warmup_eta = 1
    size_t o = (size_t)b * 4096 + s0 + row;
    out[EN_OFF + o] = en;
    out[GT_OFF + o] = gt;
    out[DR_OFF + o] = dr;
    out[RS_OFF + o] = sp;
  }
}

// ---------------------------------------------------------------------------
// Top-8 per batch row + mask + gather (jax.lax.top_k tie-break: lowest index)
// ---------------------------------------------------------------------------
__global__ void topk_kernel(const float* __restrict__ slot_states,
                            float* __restrict__ out) {
  __shared__ float redV[256];
  __shared__ int   redI[256];
  __shared__ int   cIdx[8];
  __shared__ float cVal[8];
  const int b = blockIdx.x, tid = threadIdx.x;
  const float* g = out + GT_OFF + (size_t)b * 4096;

  for (int k = 0; k < 8; ++k) {
    float bv = -3.402823466e38f;
    int   bi = 0x7fffffff;
    for (int j = tid; j < 4096; j += 256) {
      bool taken = false;
      for (int m = 0; m < k; ++m) taken |= (cIdx[m] == j);
      if (taken) continue;
      float v = g[j];
      if (v > bv || (v == bv && j < bi)) { bv = v; bi = j; }
    }
    redV[tid] = bv; redI[tid] = bi;
    __syncthreads();
    for (int off = 128; off > 0; off >>= 1) {
      if (tid < off) {
        float ov = redV[tid + off]; int oi = redI[tid + off];
        if (ov > redV[tid] || (ov == redV[tid] && oi < redI[tid])) {
          redV[tid] = ov; redI[tid] = oi;
        }
      }
      __syncthreads();
    }
    if (tid == 0) { cIdx[k] = redI[0]; cVal[k] = redV[0]; }
    __syncthreads();
  }

  if (tid < 8) {
    out[IDX_OFF + b * 8 + tid] = (float)cIdx[tid];
    out[SC_OFF  + b * 8 + tid] = cVal[tid];
  }
  for (int j = tid; j < 4096; j += 256) {
    bool m = false;
    for (int t = 0; t < 8; ++t) m |= (cIdx[t] == j);
    out[MASK_OFF + (size_t)b * 4096 + j] = m ? 1.f : 0.f;
  }
  for (int t = tid; t < 1024; t += 256) {
    int k = t >> 7, col = t & 127;
    out[AS_OFF + (size_t)b * 1024 + t] =
        slot_states[((size_t)b * 4096 + cIdx[k]) * 128 + col];
  }
}

// ---------------------------------------------------------------------------
extern "C" void kernel_launch(void* const* d_in, const int* in_sizes, int n_in,
                              void* d_out, int out_size, void* d_ws, size_t ws_size,
                              hipStream_t stream) {
  (void)in_sizes; (void)n_in; (void)out_size; (void)ws_size;
  const float* event = (const float*)d_in[0];
  const float* slot  = (const float*)d_in[1];
  const float* ep_W  = (const float*)d_in[2];
  const float* ep_b  = (const float*)d_in[3];
  const float* sp_W  = (const float*)d_in[4];
  const float* sp_b  = (const float*)d_in[5];
  const float* ln_g  = (const float*)d_in[6];
  const float* ln_b  = (const float*)d_in[7];
  const float* d_W1  = (const float*)d_in[8];
  const float* d_b1  = (const float*)d_in[9];
  const float* d_W2  = (const float*)d_in[10];
  const float* d_b2  = (const float*)d_in[11];
  const float* r_W1  = (const float*)d_in[12];
  const float* r_b1  = (const float*)d_in[13];
  const float* r_W2  = (const float*)d_in[14];
  const float* r_b2  = (const float*)d_in[15];
  const float* gbias = (const float*)d_in[16];
  float* out = (float*)d_out;

  // workspace: [spW bf16 32KB][W1cat bf16 384KB][ef f32 32KB]
  bf16_t* wSpW = (bf16_t*)d_ws;
  bf16_t* wW1  = wSpW + 16384;
  float*  wEF  = (float*)((char*)d_ws + 32768 + 393216);

  cvt_f32_bf16<<<64, 256, 0, stream>>>(sp_W, wSpW, 16384);
  cvt_f32_bf16<<<384, 256, 0, stream>>>(d_W1, wW1, 98304);
  cvt_f32_bf16<<<384, 256, 0, stream>>>(r_W1, wW1 + 98304, 98304);
  event_proj_kernel<<<32, 256, 0, stream>>>(event, ep_W, ep_b, wEF);
  eml_main_kernel<<<8192, 256, 0, stream>>>(slot, sp_b, ln_g, ln_b,
      d_b1, d_W2, d_b2, r_b1, r_W2, r_b2, gbias, wSpW, wW1, wEF, out);
  topk_kernel<<<64, 256, 0, stream>>>(slot, out);
}
